// PositionwiseFeedForwardUttMoE_29841432773031
// MI455X (gfx1250) — matrix operand order repaired and checked by reading, not compiled, not run on hardware
//
#include <hip/hip_runtime.h>
#include <cstdint>

#define B_ 32
#define T_ 512
#define D_ 512
#define E_ 8
#define H_ 2048

#define BM 128
#define BN 128
#define BK 64
#define LDAe 72     // padded LDS stride in bf16 elems (144 B rows, 16B-aligned chunks)

typedef __attribute__((ext_vector_type(16))) __bf16 bf16x16;
typedef __attribute__((ext_vector_type(8)))  __bf16 bf16x8;
typedef __attribute__((ext_vector_type(8)))  float  f32x8;
typedef __attribute__((ext_vector_type(4)))  unsigned short u16x4;

union FragAB { bf16x16 v; bf16x8 h[2]; };

__device__ __forceinline__ unsigned short f2bf_bits(float f) {
    union { float f; unsigned u; } c; c.f = f;
    unsigned u = c.u;
    unsigned r = u + 0x7FFFu + ((u >> 16) & 1u);   // round-to-nearest-even
    return (unsigned short)(r >> 16);
}

// ---------------- Router: mean-pool -> softmax -> argmax ----------------
__global__ void router_kernel(const float* __restrict__ x,
                              const float* __restrict__ Wp,
                              const float* __restrict__ bp,
                              float* __restrict__ probs_out,
                              float* __restrict__ expert_out,
                              int* __restrict__ expert_ws) {
    const int b = blockIdx.x;
    const int tid = threadIdx.x;          // 256 threads
    __shared__ float pooled[D_];
    __shared__ float logits[E_];
    const float* xb = x + (size_t)b * T_ * D_;
    for (int d = tid; d < D_; d += 256) {
        float s = 0.f;
        for (int t = 0; t < T_; ++t) s += xb[(size_t)t * D_ + d];
        pooled[d] = s * (1.0f / (float)T_);
    }
    __syncthreads();
    if (tid < E_) {
        float acc = bp[tid];
        for (int d = 0; d < D_; ++d) acc += pooled[d] * Wp[d * E_ + tid];
        logits[tid] = acc;
    }
    __syncthreads();
    if (tid == 0) {
        float m = logits[0]; int arg = 0;
        for (int e = 1; e < E_; ++e) if (logits[e] > m) { m = logits[e]; arg = e; }
        float p[E_]; float sum = 0.f;
        for (int e = 0; e < E_; ++e) { p[e] = __expf(logits[e] - m); sum += p[e]; }
        const float inv = 1.0f / sum;
        for (int e = 0; e < E_; ++e) probs_out[b * E_ + e] = p[e] * inv;
        expert_out[b] = (float)arg;
        expert_ws[b]  = arg;
    }
}

// ---------------- Elementwise f32 -> bf16 (vec4) ------------------------
__global__ __launch_bounds__(256) void cvt_x_kernel(const float* __restrict__ in,
                                                    unsigned short* __restrict__ out,
                                                    int n4) {
    int i = blockIdx.x * 256 + threadIdx.x;
    if (i >= n4) return;
    const float4 v = ((const float4*)in)[i];
    u16x4 o;
    o.x = f2bf_bits(v.x); o.y = f2bf_bits(v.y);
    o.z = f2bf_bits(v.z); o.w = f2bf_bits(v.w);
    ((u16x4*)out)[i] = o;
}

// ---------------- Tiled transpose + convert: in[R,C] f32 -> out[C,R] bf16
__global__ __launch_bounds__(256) void transpose_cvt_kernel(
    const float* __restrict__ in, unsigned short* __restrict__ out, int R, int C) {
    __shared__ unsigned short tile[32][33];
    const float* src = in + (size_t)blockIdx.z * R * C;
    unsigned short* dst = out + (size_t)blockIdx.z * R * C;
    const int c0 = blockIdx.x * 32, r0 = blockIdx.y * 32;
    const int tx = threadIdx.x & 31, ty = threadIdx.x >> 5;   // 32 x 8
    #pragma unroll
    for (int i = 0; i < 32; i += 8)
        tile[ty + i][tx] = f2bf_bits(src[(size_t)(r0 + ty + i) * C + (c0 + tx)]);
    __syncthreads();
    #pragma unroll
    for (int i = 0; i < 32; i += 8)
        dst[(size_t)(c0 + ty + i) * R + (r0 + tx)] = tile[tx][ty + i];
}

// ---------------- Async tile issue: A[128xBK] + Bt[128xBK] bf16 rows ----
// Both operand matrices are row-major with row length K (A: m-rows, Bt: n-rows).
// BK=64 -> 128-byte rows -> 8x16B chunks/row -> 4 chunks per thread per matrix.
// Each thread issues 8 ASYNCcnt ops per tile-pair.
__device__ __forceinline__ void issue_tile(unsigned long long abase,
                                           unsigned long long bbase,
                                           int K, int k0,
                                           unsigned asoff, unsigned bsoff,
                                           int tid) {
    #pragma unroll
    for (int q = 0; q < 4; ++q) {
        const int ch   = tid + q * 256;        // 0..1023
        const int row  = ch >> 3;              // 0..127
        const int part = ch & 7;               // 16B chunk within 128B row
        const unsigned goff = (unsigned)((row * K + k0 + part * 8) * 2);
        const unsigned loff = (unsigned)((row * LDAe + part * 8) * 2);
        const unsigned lds_a = asoff + loff;
        const unsigned lds_b = bsoff + loff;
        asm volatile("global_load_async_to_lds_b128 %0, %1, %2"
                     :: "v"(lds_a), "v"(goff), "s"(abase) : "memory");
        asm volatile("global_load_async_to_lds_b128 %0, %1, %2"
                     :: "v"(lds_b), "v"(goff), "s"(bbase) : "memory");
    }
}

// ---------------- Shared GEMM core (wave tile 32x64, BK=64, 2-buffered) -
// Computes acc[2][4] for this wave; caller handles epilogue.
__device__ __forceinline__ void gemm_core(unsigned long long abase,
                                          unsigned long long bbase,
                                          int K,
                                          unsigned short* As, unsigned short* Bs,
                                          f32x8 (&acc)[2][4]) {
    const int tid  = threadIdx.x;
    const int lane = tid & 31;
    const int wave = tid >> 5;
    const int wm = wave & 3;        // M sub-block: 32 rows
    const int wn = wave >> 2;       // N sub-block: 64 cols
    const int rowM = lane & 15;
    const int ksel = lane >> 4;

    const unsigned asbase = (unsigned)(uintptr_t)As;
    const unsigned bsbase = (unsigned)(uintptr_t)Bs;

    issue_tile(abase, bbase, K, 0, asbase, bsbase, tid);

    const int NIT = K / BK;
    for (int it = 0; it < NIT; ++it) {
        const int cur = it & 1;
        if (it + 1 < NIT) {
            issue_tile(abase, bbase, K, (it + 1) * BK,
                       asbase + (unsigned)((cur ^ 1) * (BM * LDAe * 2)),
                       bsbase + (unsigned)((cur ^ 1) * (BN * LDAe * 2)), tid);
            asm volatile("s_wait_asynccnt 0x8" ::: "memory");
        } else {
            asm volatile("s_wait_asynccnt 0x0" ::: "memory");
        }
        __syncthreads();

        const unsigned short* Ab = As + cur * (BM * LDAe);
        const unsigned short* Bb = Bs + cur * (BN * LDAe);
        #pragma unroll
        for (int ks = 0; ks < 2; ++ks) {
            const int ko = ks * 32;
            FragAB a[2];
            #pragma unroll
            for (int mi = 0; mi < 2; ++mi) {
                const int arow = wm * 32 + mi * 16 + rowM;
                a[mi].h[0] = *(const bf16x8*)(Ab + arow * LDAe + ko + ksel * 8);
                a[mi].h[1] = *(const bf16x8*)(Ab + arow * LDAe + ko + ksel * 8 + 16);
            }
            FragAB bf[4];
            #pragma unroll
            for (int ji = 0; ji < 4; ++ji) {
                const int bcol = wn * 64 + ji * 16 + rowM;
                bf[ji].h[0] = *(const bf16x8*)(Bb + bcol * LDAe + ko + ksel * 8);
                bf[ji].h[1] = *(const bf16x8*)(Bb + bcol * LDAe + ko + ksel * 8 + 16);
            }
            #pragma unroll
            for (int mi = 0; mi < 2; ++mi)
                #pragma unroll
                for (int ji = 0; ji < 4; ++ji)
                    acc[mi][ji] = __builtin_amdgcn_wmma_f32_16x16x32_bf16(
                        false, a[mi].v, false, bf[ji].v, (short)0, acc[mi][ji],
                        false, false);
        }
        __syncthreads();
    }
}

// ---------------- FFN layer 1: h = relu(xbf @ W1t^T + b1[e]) (bf16 out) -
__global__ __launch_bounds__(256) void ffn1_kernel(
    const unsigned short* __restrict__ xbf,   // [B,T,D] bf16
    const unsigned short* __restrict__ W1t,   // [E,H,D] bf16 (transposed)
    const float* __restrict__ b1,             // [E,H]
    const int*   __restrict__ expert,         // [B]
    unsigned short* __restrict__ h_out)       // [B,T,H] bf16
{
    __shared__ alignas(16) unsigned short As[2 * BM * LDAe];
    __shared__ alignas(16) unsigned short Bs[2 * BN * LDAe];

    const int bm = blockIdx.x, bn = blockIdx.y, b = blockIdx.z;
    const int e  = expert[b];

    const unsigned long long abase = (unsigned long long)(uintptr_t)
        (xbf + (size_t)b * T_ * D_ + (size_t)(bm * BM) * D_);
    const unsigned long long bbase = (unsigned long long)(uintptr_t)
        (W1t + (size_t)e * H_ * D_ + (size_t)(bn * BN) * D_);

    f32x8 acc[2][4];
    #pragma unroll
    for (int mi = 0; mi < 2; ++mi)
        #pragma unroll
        for (int ji = 0; ji < 4; ++ji)
            #pragma unroll
            for (int r = 0; r < 8; ++r) acc[mi][ji][r] = 0.f;

    gemm_core(abase, bbase, D_, As, Bs, acc);

    const int lane = threadIdx.x & 31;
    const int wave = threadIdx.x >> 5;
    const int wm = wave & 3, wn = wave >> 2;
    const int rowHalf = (lane >> 4) << 3;
    const int colLane = lane & 15;
    unsigned short* Hb = h_out + (size_t)b * T_ * H_;
    #pragma unroll
    for (int mi = 0; mi < 2; ++mi) {
        const int rowBase = bm * BM + wm * 32 + mi * 16 + rowHalf;
        #pragma unroll
        for (int ji = 0; ji < 4; ++ji) {
            const int col = bn * BN + wn * 64 + ji * 16 + colLane;
            const float bias = b1[e * H_ + col];
            #pragma unroll
            for (int r = 0; r < 8; ++r) {
                float v = acc[mi][ji][r] + bias;
                v = v > 0.f ? v : 0.f;
                Hb[(size_t)(rowBase + r) * H_ + col] = f2bf_bits(v);
            }
        }
    }
}

// ---------------- FFN layer 2: out = h @ W2t^T + b2[e] (f32 out) --------
__global__ __launch_bounds__(256) void ffn2_kernel(
    const unsigned short* __restrict__ h_in,  // [B,T,H] bf16
    const unsigned short* __restrict__ W2t,   // [E,D,H] bf16 (transposed)
    const float* __restrict__ b2,             // [E,D]
    const int*   __restrict__ expert,         // [B]
    float* __restrict__ out)                  // [B,T,D] f32
{
    __shared__ alignas(16) unsigned short As[2 * BM * LDAe];
    __shared__ alignas(16) unsigned short Bs[2 * BN * LDAe];

    const int bm = blockIdx.x, bn = blockIdx.y, b = blockIdx.z;
    const int e  = expert[b];

    const unsigned long long abase = (unsigned long long)(uintptr_t)
        (h_in + (size_t)b * T_ * H_ + (size_t)(bm * BM) * H_);
    const unsigned long long bbase = (unsigned long long)(uintptr_t)
        (W2t + (size_t)e * D_ * H_ + (size_t)(bn * BN) * H_);

    f32x8 acc[2][4];
    #pragma unroll
    for (int mi = 0; mi < 2; ++mi)
        #pragma unroll
        for (int ji = 0; ji < 4; ++ji)
            #pragma unroll
            for (int r = 0; r < 8; ++r) acc[mi][ji][r] = 0.f;

    gemm_core(abase, bbase, H_, As, Bs, acc);

    const int lane = threadIdx.x & 31;
    const int wave = threadIdx.x >> 5;
    const int wm = wave & 3, wn = wave >> 2;
    const int rowHalf = (lane >> 4) << 3;
    const int colLane = lane & 15;
    float* Ob = out + (size_t)b * T_ * D_;
    #pragma unroll
    for (int mi = 0; mi < 2; ++mi) {
        const int rowBase = bm * BM + wm * 32 + mi * 16 + rowHalf;
        #pragma unroll
        for (int ji = 0; ji < 4; ++ji) {
            const int col = bn * BN + wn * 64 + ji * 16 + colLane;
            const float bias = b2[e * D_ + col];
            #pragma unroll
            for (int r = 0; r < 8; ++r)
                Ob[(size_t)(rowBase + r) * D_ + col] = acc[mi][ji][r] + bias;
        }
    }
}

extern "C" void kernel_launch(void* const* d_in, const int* in_sizes, int n_in,
                              void* d_out, int out_size, void* d_ws, size_t ws_size,
                              hipStream_t stream) {
    (void)in_sizes; (void)n_in; (void)out_size; (void)ws_size;
    const float* x  = (const float*)d_in[0];
    const float* Wp = (const float*)d_in[1];
    const float* bp = (const float*)d_in[2];
    const float* W1 = (const float*)d_in[3];
    const float* b1 = (const float*)d_in[4];
    const float* W2 = (const float*)d_in[5];
    const float* b2 = (const float*)d_in[6];

    float* out        = (float*)d_out;
    float* probs_out  = out + (size_t)B_ * T_ * D_;       // [B,E]
    float* expert_out = probs_out + B_ * E_;              // [B,1]

    char* ws = (char*)d_ws;
    int* expert_ws        = (int*)ws;                                    // 32 ints
    unsigned short* xbf   = (unsigned short*)(ws + 256);                 // [B,T,D]
    unsigned short* W1t   = xbf + (size_t)B_ * T_ * D_;                  // [E,H,D]
    unsigned short* W2t   = W1t + (size_t)E_ * D_ * H_;                  // [E,D,H]
    unsigned short* hws   = W2t + (size_t)E_ * H_ * D_;                  // [B,T,H]

    router_kernel<<<B_, 256, 0, stream>>>(x, Wp, bp, probs_out, expert_out, expert_ws);

    const int n4 = (B_ * T_ * D_) / 4;
    cvt_x_kernel<<<(n4 + 255) / 256, 256, 0, stream>>>(x, xbf, n4);

    transpose_cvt_kernel<<<dim3(H_ / 32, D_ / 32, E_), 256, 0, stream>>>(W1, W1t, D_, H_);
    transpose_cvt_kernel<<<dim3(D_ / 32, H_ / 32, E_), 256, 0, stream>>>(W2, W2t, H_, D_);

    dim3 g1(T_ / BM, H_ / BN, B_);   // 4 x 16 x 32
    ffn1_kernel<<<g1, 256, 0, stream>>>(xbf, W1t, b1, expert_ws, hws);

    dim3 g2(T_ / BM, D_ / BN, B_);   // 4 x 4 x 32
    ffn2_kernel<<<g2, 256, 0, stream>>>(hws, W2t, b2, expert_ws, out);
}
